// PixConsistLoss_54657753808986
// MI455X (gfx1250) — matrix operand order repaired and checked by reading, not compiled, not run on hardware
//
#include <hip/hip_runtime.h>
#include <hip/hip_bf16.h>

typedef __attribute__((ext_vector_type(16))) _Float16 v16h;
typedef __attribute__((ext_vector_type(8)))  float    v8f;

#define NPIX 2304   // 48*48
#define CCH  256
#define BATCH 32
#define BM 128
#define BN 128
#define LDK 72      // LDS row stride in halves (64 K + 8 pad) -> 144B rows, 16B aligned

__global__ void pcl_zero_kernel(float* p, int n) {
    int i = blockIdx.x * blockDim.x + threadIdx.x;
    if (i < n) p[i] = 0.0f;
}

// Per-pixel squared norms over channels; coalesced along pixel dim.
__global__ __launch_bounds__(256) void pcl_norms_kernel(
    const float* __restrict__ x1, const float* __restrict__ x2,
    float* __restrict__ ss1, float* __restrict__ ss2) {
    int idx = blockIdx.x * 256 + threadIdx.x;       // 0 .. B*NPIX-1
    int b = idx / NPIX;
    int n = idx - b * NPIX;
    const float* p1 = x1 + (size_t)b * CCH * NPIX + n;
    const float* p2 = x2 + (size_t)b * CCH * NPIX + n;
    float s1 = 0.f, s2 = 0.f;
    for (int c = 0; c < CCH; ++c) {
        float v1 = p1[(size_t)c * NPIX];
        float v2 = p2[(size_t)c * NPIX];
        s1 += v1 * v1;
        s2 += v2 * v2;
    }
    ss1[idx] = s1;
    ss2[idx] = s2;
}

// Tiled WMMA GEMM (M=N=2304, K=256) fused with cosine-normalize + masked reduction.
__global__ __launch_bounds__(256) void pcl_gemm_mask_kernel(
    const float* __restrict__ x1, const float* __restrict__ x2,
    const unsigned char* __restrict__ mask,
    const float* __restrict__ ss1, const float* __restrict__ ss2,
    float* __restrict__ result, float* __restrict__ matches) {

    __shared__ __align__(16) _Float16 As[BM * LDK];
    __shared__ __align__(16) _Float16 Bs[BN * LDK];
    __shared__ float red[16];

    const int tid  = threadIdx.x;
    const int lane = tid & 31;
    const int wave = tid >> 5;
    const int b  = blockIdx.z;
    const int m0 = blockIdx.y * BM;   // rows   <- x1 pixels
    const int n0 = blockIdx.x * BN;   // cols   <- x2 pixels

    // Prefetch this block's 128x128 mask tile (overlaps with GEMM).
    {
        const unsigned char* mt = mask + (size_t)b * NPIX * NPIX
                                + (size_t)(m0 + (tid >> 1)) * NPIX + n0 + (tid & 1) * 64;
        __builtin_prefetch(mt, 0, 1);
    }

    v8f z;
    #pragma unroll
    for (int e = 0; e < 8; ++e) z[e] = 0.f;
    v8f acc[2][4];
    #pragma unroll
    for (int mi = 0; mi < 2; ++mi)
        #pragma unroll
        for (int ni = 0; ni < 4; ++ni) acc[mi][ni] = z;

    // Each thread owns one LDS row (one pixel): threads 0..127 -> A, 128..255 -> B.
    const int row = tid & 127;
    const float* gsrc = (tid < 128)
        ? (x1 + (size_t)b * CCH * NPIX + m0 + row)
        : (x2 + (size_t)b * CCH * NPIX + n0 + row);
    _Float16* lrow = ((tid < 128) ? As : Bs) + row * LDK;

    const int wm = wave >> 1;           // 0..3  -> 32-row wave tile
    const int wn = wave & 1;            // 0..1  -> 64-col wave tile
    const _Float16* aw = As + (wm * 32) * LDK;
    const _Float16* bw = Bs + (wn * 64) * LDK;
    const int frow = lane & 15;
    const int asel = (lane >> 4) * 8;   // A-frag K sub-offset (ISA 16-bit A 16x32 layout)
    const int bsel = (lane >> 4) * 8;   // B-frag K sub-offset (same table order as A)

    for (int kb = 0; kb < CCH; kb += 64) {
        __syncthreads();
        const float* gp = gsrc + (size_t)kb * NPIX;
        #pragma unroll
        for (int g = 0; g < 8; ++g) {
            union { _Float16 h[8]; uint4 q; } pk;
            #pragma unroll
            for (int j = 0; j < 8; ++j)
                pk.h[j] = (_Float16)gp[(size_t)(g * 8 + j) * NPIX];
            *(uint4*)(lrow + g * 8) = pk.q;   // K-contiguous f16 row, 16B aligned
        }
        __syncthreads();

        #pragma unroll
        for (int kk = 0; kk < 64; kk += 32) {
            v16h af[2];
            v16h bf[4];
            #pragma unroll
            for (int mi = 0; mi < 2; ++mi) {
                const _Float16* p = aw + (mi * 16 + frow) * LDK + kk + asel;
                union { v16h v; uint4 q[2]; } u;
                u.q[0] = *(const uint4*)(p);        // K 0..7  (or 8..15 for hi lanes)
                u.q[1] = *(const uint4*)(p + 16);   // K 16..23 (or 24..31)
                af[mi] = u.v;
            }
            #pragma unroll
            for (int ni = 0; ni < 4; ++ni) {
                const _Float16* p = bw + (ni * 16 + frow) * LDK + kk + bsel;
                union { v16h v; uint4 q[2]; } u;
                u.q[0] = *(const uint4*)(p);        // K 0..7  (or 8..15 for hi lanes)
                u.q[1] = *(const uint4*)(p + 16);   // K 16..23 (or 24..31)
                bf[ni] = u.v;
            }
            #pragma unroll
            for (int mi = 0; mi < 2; ++mi)
                #pragma unroll
                for (int ni = 0; ni < 4; ++ni)
                    acc[mi][ni] = __builtin_amdgcn_wmma_f32_16x16x32_f16(
                        false, af[mi], false, bf[ni],
                        (short)0, acc[mi][ni], false, false);
        }
    }

    // Epilogue: cosine normalize, mask, reduce. C/D layout: VGPR r, lanes0-15 M=r,
    // lanes16-31 M=8+r; N = lane&15.
    float lsum = 0.f, lcnt = 0.f;
    const int mlane = (lane >> 4) * 8;

    float rn2c[4];
    #pragma unroll
    for (int ni = 0; ni < 4; ++ni) {
        int n = n0 + wn * 64 + ni * 16 + frow;
        rn2c[ni] = sqrtf(ss2[b * NPIX + n]);
    }

    #pragma unroll
    for (int mi = 0; mi < 2; ++mi) {
        #pragma unroll
        for (int r = 0; r < 8; ++r) {
            int m = m0 + wm * 32 + mi * 16 + mlane + r;
            float n1 = sqrtf(ss1[b * NPIX + m]);
            const unsigned char* mrow = mask + (size_t)b * NPIX * NPIX + (size_t)m * NPIX;
            #pragma unroll
            for (int ni = 0; ni < 4; ++ni) {
                int n = n0 + wn * 64 + ni * 16 + frow;
                unsigned char mk = mrow[n];
                if (mk) {
                    float denom = fmaxf(n1 * rn2c[ni], 1e-8f);
                    lsum += acc[mi][ni][r] / denom;
                    lcnt += 1.f;
                }
            }
        }
    }

    #pragma unroll
    for (int off = 16; off; off >>= 1) {
        lsum += __shfl_xor(lsum, off);
        lcnt += __shfl_xor(lcnt, off);
    }
    if (lane == 0) { red[wave] = lsum; red[8 + wave] = lcnt; }
    __syncthreads();
    if (tid == 0) {
        float ts = 0.f, tc = 0.f;
        #pragma unroll
        for (int w = 0; w < 8; ++w) { ts += red[w]; tc += red[8 + w]; }
        atomicAdd(&result[b], ts);
        atomicAdd(&matches[b], tc);
    }
}

__global__ void pcl_final_kernel(const float* __restrict__ result,
                                 const float* __restrict__ matches,
                                 float* __restrict__ out) {
    int bidx = threadIdx.x;   // 32 threads = one wave
    float r = result[bidx];
    float m = matches[bidx];
    bool valid = m > 0.f;
    float per = valid ? r / fmaxf(m, 1.f) : 0.f;
    float v = valid ? 1.f : 0.f;
    #pragma unroll
    for (int off = 16; off; off >>= 1) {
        per += __shfl_down(per, off);
        v   += __shfl_down(v, off);
    }
    if (bidx == 0) out[0] = 1.f - per / fmaxf(v, 1.f);
}

extern "C" void kernel_launch(void* const* d_in, const int* in_sizes, int n_in,
                              void* d_out, int out_size, void* d_ws, size_t ws_size,
                              hipStream_t stream) {
    const float* x1 = (const float*)d_in[0];
    const float* x2 = (const float*)d_in[1];
    const unsigned char* mask = (const unsigned char*)d_in[2];

    float* ws      = (float*)d_ws;
    float* ss1     = ws;                         // B*NPIX floats
    float* ss2     = ws + BATCH * NPIX;          // B*NPIX floats
    float* result  = ws + 2 * BATCH * NPIX;      // 32 floats
    float* matches = result + BATCH;             // 32 floats

    pcl_zero_kernel<<<1, 64, 0, stream>>>(result, 2 * BATCH);

    pcl_norms_kernel<<<(BATCH * NPIX) / 256, 256, 0, stream>>>(x1, x2, ss1, ss2);

    dim3 grid(NPIX / BN, NPIX / BM, BATCH);      // (18, 18, 32)
    pcl_gemm_mask_kernel<<<grid, 256, 0, stream>>>(x1, x2, mask, ss1, ss2,
                                                   result, matches);

    pcl_final_kernel<<<1, 32, 0, stream>>>(result, matches, (float*)d_out);
}
